// HunyuanVideoMidBlock3D_47287589929599
// MI455X (gfx1250) — compile-verified, matrix operand order checked by reading, not run on hardware
//
#include <hip/hip_runtime.h>
#include <hip/hip_bf16.h>
#include <math.h>

// ---------------------------------------------------------------------------
// HunyuanVideo mid-block on MI455X (gfx1250, wave32, WMMA bf16 w/ f32 accum)
// (S, C) row-major working layout, S = 9216, C = 512.
// Per-wave 16x64 tiles, 4 accumulators, 2-deep software pipeline enforced by
// a data-tied asm scheduling fence: [loads kk+1] [fence(acc)] [wmmas kk].
// ---------------------------------------------------------------------------

#define CCH 512
#define TFR 9
#define HDI 32
#define WDI 32
#define SPOS (TFR*HDI*WDI)      // 9216
#define NGRP 32
#define CPG  (CCH/NGRP)         // 16
#define NELT (SPOS*CCH)         // 4718592

typedef __attribute__((ext_vector_type(16))) __bf16 v16bf;
typedef __attribute__((ext_vector_type(8)))  float  v8f;

__device__ __forceinline__ unsigned short f2bf(float f) {
  unsigned int u = __float_as_uint(f);
  u += 0x7fffu + ((u >> 16) & 1u);           // round-to-nearest-even
  return (unsigned short)(u >> 16);
}

// Build a 16x32 bf16 WMMA fragment from a row-major [row][k] slice.
// Per ISA 7.12.2 (16-bit A-matrix 16x32): lane half h=lane/16 holds
// K in {8h..8h+7} U {16+8h..16+8h+7}; pairs contiguous -> two b128 loads.
__device__ __forceinline__ v16bf load_frag(const unsigned short* p, int h) {
  union { v16bf v; unsigned int u[8]; } f;
  const int base = 8 * h;
  #pragma unroll
  for (int i = 0; i < 4; ++i)
    f.u[i] = *(const unsigned int*)(p + base + 2 * i);
  #pragma unroll
  for (int i = 0; i < 4; ++i)
    f.u[4 + i] = *(const unsigned int*)(p + base + 16 + 2 * i);
  return f.v;
}

__device__ __forceinline__ v8f wmma_bf16(v16bf a, v16bf b, v8f c) {
  return __builtin_amdgcn_wmma_f32_16x16x32_bf16(false, a, false, b,
                                                 (short)0, c, false, false);
}

// Scheduling fence: memory ops above cannot sink past it; instructions that
// consume the tied accumulators cannot hoist above it.
#define FENCE4(a0_, a1_, a2_, a3_) \
  asm volatile("" : "+v"(a0_), "+v"(a1_), "+v"(a2_), "+v"(a3_) :: "memory")
#define FENCE2(a0_, a1_) \
  asm volatile("" : "+v"(a0_), "+v"(a1_) :: "memory")
#define FENCE8(a0_, a1_, a2_, a3_, a4_, a5_, a6_, a7_) \
  asm volatile("" : "+v"(a0_), "+v"(a1_), "+v"(a2_), "+v"(a3_), \
                    "+v"(a4_), "+v"(a5_), "+v"(a6_), "+v"(a7_) :: "memory")

// ---------------------------------------------------------------------------
// Layout shuffles
// ---------------------------------------------------------------------------

__global__ void k_nchw_to_sc(const float* __restrict__ in, float* __restrict__ out) {
  int idx = blockIdx.x * blockDim.x + threadIdx.x;
  if (idx >= NELT) return;
  int c = idx & (CCH - 1);
  int s = idx >> 9;
  out[idx] = in[(size_t)c * SPOS + s];
}

__global__ void k_sc_to_nchw(const float* __restrict__ in, float* __restrict__ out) {
  int idx = blockIdx.x * blockDim.x + threadIdx.x;
  if (idx >= NELT) return;
  int s = idx % SPOS;
  int c = idx / SPOS;
  out[idx] = in[(size_t)s * CCH + c];
}

// conv weight f32 [O][I][3][3][3] -> bf16 [tap][O][I]
__global__ void k_repack_conv_w(const float* __restrict__ w, unsigned short* __restrict__ out) {
  int idx = blockIdx.x * blockDim.x + threadIdx.x;
  if (idx >= 27 * CCH * CCH) return;
  int tap = idx / (CCH * CCH);
  int rem = idx - tap * CCH * CCH;
  int o = rem >> 9;
  int i = rem & (CCH - 1);
  out[idx] = f2bf(w[((size_t)o * CCH + i) * 27 + tap]);
}

// attn weight f32 [K][N] -> bf16 [N][K]
__global__ void k_repack_attn_w(const float* __restrict__ w, unsigned short* __restrict__ out) {
  int idx = blockIdx.x * blockDim.x + threadIdx.x;
  if (idx >= CCH * CCH) return;
  int n = idx >> 9;
  int k = idx & (CCH - 1);
  out[idx] = f2bf(w[(size_t)k * CCH + n]);
}

// ---------------------------------------------------------------------------
// GroupNorm
// ---------------------------------------------------------------------------
__global__ void k_gn_stats(const float* __restrict__ X, float* __restrict__ stats) {
  int g = blockIdx.x;
  int tid = threadIdx.x;
  float s = 0.f, s2 = 0.f;
  const float* base = X + g * CPG;
  for (int i = tid; i < SPOS; i += blockDim.x) {
    const float* p = base + (size_t)i * CCH;
    #pragma unroll
    for (int j = 0; j < CPG; ++j) { float v = p[j]; s += v; s2 += v * v; }
  }
  __shared__ float sh[256], sh2[256];
  sh[tid] = s; sh2[tid] = s2;
  __syncthreads();
  for (int o = 128; o > 0; o >>= 1) {
    if (tid < o) { sh[tid] += sh[tid + o]; sh2[tid] += sh2[tid + o]; }
    __syncthreads();
  }
  if (tid == 0) {
    float n = (float)(SPOS * CPG);
    float mu = sh[0] / n;
    float var = sh2[0] / n - mu * mu;
    stats[g * 2]     = mu;
    stats[g * 2 + 1] = rsqrtf(var + 1e-6f);
  }
}

__global__ void k_gn_apply(const float* __restrict__ X, const float* __restrict__ stats,
                           const float* __restrict__ gamma, const float* __restrict__ beta,
                           unsigned short* __restrict__ Yb, int do_silu) {
  int idx = blockIdx.x * blockDim.x + threadIdx.x;
  if (idx >= NELT) return;
  int c = idx & (CCH - 1);
  int g = c >> 4;
  float v = (X[idx] - stats[g * 2]) * stats[g * 2 + 1] * gamma[c] + beta[c];
  if (do_silu) v = v / (1.f + __expf(-v));
  Yb[idx] = f2bf(v);
}

// ---------------------------------------------------------------------------
// Causal conv3d as implicit GEMM.  Per-wave 16x64 tile, 27 taps x 16 K-steps,
// 2-deep pipeline with asm fence inside each tap.
// ---------------------------------------------------------------------------
__global__ void k_conv_igemm(const unsigned short* __restrict__ Xb,
                             const unsigned short* __restrict__ Wb,  // [27][512][512] bf16
                             const float* __restrict__ bias,
                             const float* __restrict__ residual,
                             float* __restrict__ Y) {
  const int mt = blockIdx.x;
  const int n0 = blockIdx.y * 64;
  const int lane = threadIdx.x;
  const int l15 = lane & 15;
  const int h = lane >> 4;

  const int row = mt * 16 + l15;             // this lane's A-fragment row
  const int t  = row >> 10;
  const int rm = row & 1023;
  const int yy = rm >> 5;
  const int xx = rm & 31;

  v8f acc[4];
  #pragma unroll
  for (int j = 0; j < 4; ++j) { v8f z = {}; acc[j] = z; }

  const unsigned short* bbase[4];
  #pragma unroll
  for (int j = 0; j < 4; ++j)
    bbase[j] = Wb + (size_t)(n0 + j * 16 + l15) * CCH;

  #pragma unroll 1
  for (int tap = 0; tap < 27; ++tap) {
    const int kt = tap / 9, khw = tap % 9, kh = khw / 3, kw = khw % 3;
    int ti = t + kt - 2;  ti = ti < 0 ? 0 : ti;                      // causal time pad (edge)
    int hi = yy + kh - 1; hi = hi < 0 ? 0 : (hi > 31 ? 31 : hi);     // spatial edge pad
    int wi = xx + kw - 1; wi = wi < 0 ? 0 : (wi > 31 ? 31 : wi);
    const unsigned short* arow = Xb + (size_t)((ti << 10) + (hi << 5) + wi) * CCH;
    const size_t toff = (size_t)tap * CCH * CCH;

    // prologue: fragments for kk = 0
    v16bf a0 = load_frag(arow, h);
    v16bf b0[4];
    #pragma unroll
    for (int j = 0; j < 4; ++j) b0[j] = load_frag(bbase[j] + toff, h);

    #pragma unroll 2
    for (int kk = 0; kk < CCH / 32 - 1; ++kk) {
      // prefetch kk+1 into fresh registers
      v16bf a1 = load_frag(arow + (kk + 1) * 32, h);
      v16bf b1[4];
      #pragma unroll
      for (int j = 0; j < 4; ++j)
        b1[j] = load_frag(bbase[j] + toff + (kk + 1) * 32, h);
      // fence: prefetch loads stay above, wmmas stay below
      FENCE4(acc[0], acc[1], acc[2], acc[3]);
      // consume kk (its loads were issued one step earlier)
      #pragma unroll
      for (int j = 0; j < 4; ++j)
        acc[j] = wmma_bf16(a0, b0[j], acc[j]);
      a0 = a1;
      #pragma unroll
      for (int j = 0; j < 4; ++j) b0[j] = b1[j];
    }
    #pragma unroll
    for (int j = 0; j < 4; ++j)
      acc[j] = wmma_bf16(a0, b0[j], acc[j]);
  }

  #pragma unroll
  for (int j = 0; j < 4; ++j) {
    const float bn = bias[n0 + j * 16 + l15];
    #pragma unroll
    for (int r = 0; r < 8; ++r) {
      int orow = mt * 16 + r + 8 * h;        // C/D layout: lane half picks M+8
      int col  = n0 + j * 16 + l15;
      float v = acc[j][r] + bn;
      if (residual) v += residual[(size_t)orow * CCH + col];
      Y[(size_t)orow * CCH + col] = v;
    }
  }
}

// ---------------------------------------------------------------------------
// Generic bf16 GEMM (S x C) @ (C x C), per-wave 16x64 tile, fenced pipeline.
// out = (A@Bt^T + bias) * scale + residual ; Bt pre-transposed [N][K].
// ---------------------------------------------------------------------------
__global__ void k_gemm(const unsigned short* __restrict__ A,
                       const unsigned short* __restrict__ Bt,
                       const float* __restrict__ bias,
                       const float* __restrict__ residual,
                       float scale,
                       float* __restrict__ outF,
                       unsigned short* __restrict__ outB,
                       unsigned short* __restrict__ outBT) {
  const int mt = blockIdx.x;
  const int n0 = blockIdx.y * 64;
  const int lane = threadIdx.x;
  const int l15 = lane & 15;
  const int h = lane >> 4;

  const unsigned short* arow = A + (size_t)(mt * 16 + l15) * CCH;
  const unsigned short* brow[4];
  #pragma unroll
  for (int j = 0; j < 4; ++j)
    brow[j] = Bt + (size_t)(n0 + j * 16 + l15) * CCH;

  v8f acc[4];
  #pragma unroll
  for (int j = 0; j < 4; ++j) { v8f z = {}; acc[j] = z; }

  v16bf a0 = load_frag(arow, h);
  v16bf b0[4];
  #pragma unroll
  for (int j = 0; j < 4; ++j) b0[j] = load_frag(brow[j], h);

  #pragma unroll 2
  for (int kk = 0; kk < CCH / 32 - 1; ++kk) {
    v16bf a1 = load_frag(arow + (kk + 1) * 32, h);
    v16bf b1[4];
    #pragma unroll
    for (int j = 0; j < 4; ++j) b1[j] = load_frag(brow[j] + (kk + 1) * 32, h);
    FENCE4(acc[0], acc[1], acc[2], acc[3]);
    #pragma unroll
    for (int j = 0; j < 4; ++j) acc[j] = wmma_bf16(a0, b0[j], acc[j]);
    a0 = a1;
    #pragma unroll
    for (int j = 0; j < 4; ++j) b0[j] = b1[j];
  }
  #pragma unroll
  for (int j = 0; j < 4; ++j) acc[j] = wmma_bf16(a0, b0[j], acc[j]);

  #pragma unroll
  for (int j = 0; j < 4; ++j) {
    const float bn = bias[n0 + j * 16 + l15];
    #pragma unroll
    for (int r = 0; r < 8; ++r) {
      int orow = mt * 16 + r + 8 * h;
      int col  = n0 + j * 16 + l15;
      float v = (acc[j][r] + bn) * scale;
      if (residual) v += residual[(size_t)orow * CCH + col];
      if (outF)  outF [(size_t)orow * CCH + col] = v;
      if (outB)  outB [(size_t)orow * CCH + col] = f2bf(v);
      if (outBT) outBT[(size_t)col * SPOS + orow] = f2bf(v);   // V stored transposed
    }
  }
}

// ---------------------------------------------------------------------------
// Frame-causal attention, flash-style online softmax.
// One wave: 16 query rows x 128 output channels; Q fragments hoisted into
// registers across the whole key loop. Frames are 1024 positions -> causal
// mask is exactly key-block aligned: just bound the key loop.
// P tile bounced through LDS to re-layout C-fragment -> A-fragment.
// ---------------------------------------------------------------------------
__global__ void k_attn(const unsigned short* __restrict__ Qb,  // pre-scaled by 1/sqrt(C)
                       const unsigned short* __restrict__ Kb,
                       const unsigned short* __restrict__ Vt,  // [C][S] bf16
                       unsigned short* __restrict__ Ob) {
  __shared__ float          sm_s[16 * 32];
  __shared__ unsigned short sm_p[16 * 32];
  __shared__ float sm_alpha[16], sm_max[16], sm_sum[16];

  const int mt = blockIdx.x;
  const int c0 = blockIdx.y * 128;
  const int lane = threadIdx.x;
  const int l15 = lane & 15;
  const int h = lane >> 4;

  const int q0 = mt * 16;
  const int qframe = q0 >> 10;
  const int nkb = (qframe + 1) * 32;   // key blocks of 32, all frames <= qframe

  if (lane < 16) { sm_max[lane] = -3.0e38f; sm_sum[lane] = 0.f; }
  __syncthreads();

  // hoist Q fragments (invariant over key loop): 16 x 8 VGPRs
  const unsigned short* qrow = Qb + (size_t)(q0 + l15) * CCH;
  v16bf qf[CCH / 32];
  #pragma unroll
  for (int kk = 0; kk < CCH / 32; ++kk) qf[kk] = load_frag(qrow + kk * 32, h);

  v8f o[8];
  #pragma unroll
  for (int t = 0; t < 8; ++t) { v8f z = {}; o[t] = z; }

  for (int kb = 0; kb < nkb; ++kb) {
    const int k0 = kb * 32;
    // ---- scores: 16 queries x 32 keys, K = 512 channels, fenced pipeline
    v8f s0 = {}, s1 = {};
    const unsigned short* k0row = Kb + (size_t)(k0 + l15) * CCH;
    const unsigned short* k1row = Kb + (size_t)(k0 + 16 + l15) * CCH;
    v16bf c0f = load_frag(k0row, h);
    v16bf c1f = load_frag(k1row, h);
    #pragma unroll 2
    for (int kk = 0; kk < CCH / 32 - 1; ++kk) {
      v16bf n0f = load_frag(k0row + (kk + 1) * 32, h);
      v16bf n1f = load_frag(k1row + (kk + 1) * 32, h);
      FENCE2(s0, s1);
      s0 = wmma_bf16(qf[kk], c0f, s0);
      s1 = wmma_bf16(qf[kk], c1f, s1);
      c0f = n0f; c1f = n1f;
    }
    s0 = wmma_bf16(qf[CCH / 32 - 1], c0f, s0);
    s1 = wmma_bf16(qf[CCH / 32 - 1], c1f, s1);

    #pragma unroll
    for (int r = 0; r < 8; ++r) {
      int rr = r + 8 * h;
      sm_s[rr * 32 + l15]      = s0[r];
      sm_s[rr * 32 + 16 + l15] = s1[r];
    }
    __syncthreads();
    // ---- online softmax bookkeeping: lane i owns query row i
    if (lane < 16) {
      float mo = sm_max[lane], mx = mo;
      for (int j = 0; j < 32; ++j) mx = fmaxf(mx, sm_s[lane * 32 + j]);
      float alpha = __expf(mo - mx);
      float sum = 0.f;
      for (int j = 0; j < 32; ++j) {
        float p = __expf(sm_s[lane * 32 + j] - mx);
        sm_p[lane * 32 + j] = f2bf(p);
        sum += p;
      }
      sm_sum[lane] = sm_sum[lane] * alpha + sum;
      sm_max[lane] = mx;
      sm_alpha[lane] = alpha;
    }
    __syncthreads();
    // ---- rescale accumulators and do P @ V (K = 32 keys):
    // load all 8 V-fragments and the P tile first, then 8 wmmas
    float al[8];
    #pragma unroll
    for (int r = 0; r < 8; ++r) al[r] = sm_alpha[r + 8 * h];
    v16bf pa = load_frag(sm_p + l15 * 32, h);
    v16bf vb[8];
    #pragma unroll
    for (int t = 0; t < 8; ++t) {
      const unsigned short* vrow = Vt + (size_t)(c0 + t * 16 + l15) * SPOS + k0;
      vb[t] = load_frag(vrow, h);
    }
    FENCE8(o[0], o[1], o[2], o[3], o[4], o[5], o[6], o[7]);
    #pragma unroll
    for (int t = 0; t < 8; ++t) {
      #pragma unroll
      for (int r = 0; r < 8; ++r) o[t][r] *= al[r];
      o[t] = wmma_bf16(pa, vb[t], o[t]);
    }
    __syncthreads();   // protect sm_s/sm_p before next block
  }

  float inv[8];
  #pragma unroll
  for (int r = 0; r < 8; ++r) inv[r] = 1.f / sm_sum[r + 8 * h];
  #pragma unroll
  for (int t = 0; t < 8; ++t)
    #pragma unroll
    for (int r = 0; r < 8; ++r) {
      int orow = q0 + r + 8 * h;
      int col  = c0 + t * 16 + l15;
      Ob[(size_t)orow * CCH + col] = f2bf(o[t][r] * inv[r]);
    }
}

// ---------------------------------------------------------------------------
// Host orchestration
// ---------------------------------------------------------------------------
extern "C" void kernel_launch(void* const* d_in, const int* in_sizes, int n_in,
                              void* d_out, int out_size, void* d_ws, size_t ws_size,
                              hipStream_t stream) {
  (void)in_sizes; (void)n_in; (void)out_size; (void)ws_size;
  // setup_inputs insertion order: x, r0{g1,b1,w1,c1,g2,b2,w2,c2},
  //                               r1{same}, attn{gg,gb,wq,bq,wk,bk,wv,bv,wo,bo}
  const float* x    = (const float*)d_in[0];
  const float* r0g1 = (const float*)d_in[1];  const float* r0b1 = (const float*)d_in[2];
  const float* r0w1 = (const float*)d_in[3];  const float* r0c1 = (const float*)d_in[4];
  const float* r0g2 = (const float*)d_in[5];  const float* r0b2 = (const float*)d_in[6];
  const float* r0w2 = (const float*)d_in[7];  const float* r0c2 = (const float*)d_in[8];
  const float* r1g1 = (const float*)d_in[9];  const float* r1b1 = (const float*)d_in[10];
  const float* r1w1 = (const float*)d_in[11]; const float* r1c1 = (const float*)d_in[12];
  const float* r1g2 = (const float*)d_in[13]; const float* r1b2 = (const float*)d_in[14];
  const float* r1w2 = (const float*)d_in[15]; const float* r1c2 = (const float*)d_in[16];
  const float* agg  = (const float*)d_in[17]; const float* agb  = (const float*)d_in[18];
  const float* awq  = (const float*)d_in[19]; const float* abq  = (const float*)d_in[20];
  const float* awk  = (const float*)d_in[21]; const float* abk  = (const float*)d_in[22];
  const float* awv  = (const float*)d_in[23]; const float* abv  = (const float*)d_in[24];
  const float* awo  = (const float*)d_in[25]; const float* abo  = (const float*)d_in[26];

  char* ws = (char*)d_ws;
  size_t off = 0;
  auto take = [&](size_t bytes) -> char* {
    char* p = ws + off;
    off += (bytes + 255) & ~(size_t)255;
    return p;
  };
  float* Xsc = (float*)take((size_t)NELT * 4);   // also reused as final f32 buffer
  float* F1  = (float*)take((size_t)NELT * 4);
  float* F2  = (float*)take((size_t)NELT * 4);
  unsigned short* Tb = (unsigned short*)take((size_t)NELT * 2);
  unsigned short* Qb = (unsigned short*)take((size_t)NELT * 2);
  unsigned short* Kb = (unsigned short*)take((size_t)NELT * 2);
  unsigned short* Vt = (unsigned short*)take((size_t)NELT * 2);
  unsigned short* Ob = (unsigned short*)take((size_t)NELT * 2);
  unsigned short* Wc = (unsigned short*)take((size_t)27 * CCH * CCH * 2);  // conv-w slot (reused)
  unsigned short* Wa = (unsigned short*)take((size_t)CCH * CCH * 2);       // attn-w slot (reused)
  float* stats = (float*)take(NGRP * 2 * 4);

  const dim3 eltG((NELT + 255) / 256), eltB(256);
  const dim3 cwG((27 * CCH * CCH + 255) / 256);
  const dim3 awG((CCH * CCH + 255) / 256);
  const dim3 tileG(SPOS / 16, CCH / 64), waveB(32);   // 16x64 per wave
  const float qscale = 1.0f / sqrtf((float)CCH);

  auto gn = [&](const float* src, const float* g, const float* b, int silu) {
    k_gn_stats<<<NGRP, 256, 0, stream>>>(src, stats);
    k_gn_apply<<<eltG, eltB, 0, stream>>>(src, stats, g, b, Tb, silu);
  };
  auto conv = [&](const float* w, const float* bias, const float* res, float* dst) {
    k_repack_conv_w<<<cwG, eltB, 0, stream>>>(w, Wc);
    k_conv_igemm<<<tileG, waveB, 0, stream>>>(Tb, Wc, bias, res, dst);
  };
  auto proj = [&](const unsigned short* a, const float* w, const float* bias, float scale,
                  const float* res, float* oF, unsigned short* oB, unsigned short* oBT) {
    k_repack_attn_w<<<awG, eltB, 0, stream>>>(w, Wa);
    k_gemm<<<tileG, waveB, 0, stream>>>(a, Wa, bias, res, scale, oF, oB, oBT);
  };

  // ---- layout in ----
  k_nchw_to_sc<<<eltG, eltB, 0, stream>>>(x, Xsc);

  // ---- resnet 0 ----
  gn(Xsc, r0g1, r0b1, 1);  conv(r0w1, r0c1, nullptr, F1);
  gn(F1,  r0g2, r0b2, 1);  conv(r0w2, r0c2, Xsc, F2);       // F2 = h after resnet0

  // ---- frame-causal attention ----
  gn(F2, agg, agb, 0);
  proj(Tb, awq, abq, qscale, nullptr, nullptr, Qb, nullptr); // Q (pre-scaled)
  proj(Tb, awk, abk, 1.0f,   nullptr, nullptr, Kb, nullptr); // K
  proj(Tb, awv, abv, 1.0f,   nullptr, nullptr, nullptr, Vt); // V (transposed)
  k_attn<<<dim3(SPOS / 16, CCH / 128), waveB, 0, stream>>>(Qb, Kb, Vt, Ob);
  proj(Ob, awo, abo, 1.0f, F2, F1, nullptr, nullptr);        // F1 = h after attention

  // ---- resnet 1 ----
  gn(F1, r1g1, r1b1, 1);  conv(r1w1, r1c1, nullptr, F2);
  gn(F2, r1g2, r1b2, 1);  conv(r1w2, r1c2, F1, Xsc);         // Xsc = final (S,C)

  // ---- layout out ----
  k_sc_to_nchw<<<eltG, eltB, 0, stream>>>(Xsc, (float*)d_out);
}
